// Decoder_21620865368368
// MI455X (gfx1250) — matrix operand order repaired and checked by reading, not compiled, not run on hardware
//
#include <hip/hip_runtime.h>
#include <math.h>

#define Bn   16
#define Tn   512
#define En   1024
#define Hn   512
#define Vn   10000
#define Ln   64
#define An   256
#define CCHn 10
#define Kw   101
#define H4   2048

typedef __attribute__((ext_vector_type(16))) __bf16 v16bf;
typedef __attribute__((ext_vector_type(8)))  float  v8f;

// ---------------- WMMA helpers (wave32, gfx1250) ----------------

__device__ __forceinline__ v8f wmma_bf16(v16bf a, v16bf b, v8f c) {
  // D = A(16x32 bf16) x B(32x16 bf16) + C(16x16 f32)
  return __builtin_amdgcn_wmma_f32_16x16x32_bf16(false, a, false, b, (short)0, c,
                                                 false, false);
}

// Packed-A fragment: 32B contiguous per lane, layout [mtile*K32 + kb][lane][16].
__device__ __forceinline__ v16bf load_a_packed(const __bf16* __restrict__ Apk,
                                               int blk, int lane) {
  return *(const v16bf*)(Apk + ((size_t)blk * 32 + lane) * 16);
}

// B fragment from pre-transposed bf16 weights Wt (N x K row-major).
// col = lane&15 ; kbase2 = kblock*32 + (lane>>4)*16 ; element j = W[kbase2+j][col].
__device__ __forceinline__ v16bf load_b_frag(const __bf16* __restrict__ Wt, int ldk,
                                             int col, int kbase2) {
  return *(const v16bf*)(Wt + (size_t)col * ldk + kbase2);
}

__device__ __forceinline__ float sigmoidf_(float x) { return 1.0f / (1.0f + expf(-x)); }

// ---------------- weight conversion: fp32 (K x N) -> bf16 (N x K) ----------------

__global__ __launch_bounds__(256) void cvt_transpose_bf16(const float* __restrict__ in,
                                                          __bf16* __restrict__ out,
                                                          int K, int N) {
  int idx = blockIdx.x * 256 + threadIdx.x;
  if (idx >= K * N) return;
  int n = idx / K;
  int k = idx - n * K;
  out[idx] = (__bf16)in[(size_t)k * N + n];
}

// -------- pack fp32 (M x K) row-major into WMMA A-fragment bf16 layout --------
// out[(((mt*K32)+kb)*32 + lane)*16 + j] with
//   row = mt*16 + (lane&15), k = kb*32 + (lane>>4)*16 + j

__global__ __launch_bounds__(256) void pack_a_bf16(const float* __restrict__ in,
                                                   __bf16* __restrict__ out,
                                                   int M, int K) {
  int idx = blockIdx.x * 256 + threadIdx.x;
  if (idx >= M * K) return;
  const int K32 = K >> 5;
  int j       = idx & 15;
  int lane    = (idx >> 4) & 31;
  int tileblk = idx >> 9;             // mt*K32 + kb
  int kb      = tileblk % K32;
  int mt      = tileblk / K32;
  int row     = mt * 16 + (lane & 15);
  int kk      = (kb << 5) + ((lane >> 4) << 4) + j;
  out[idx] = (__bf16)in[(size_t)row * K + kk];
}

// ---------------- init: zero s, c, alpha ----------------

__global__ __launch_bounds__(256) void init_state(float* __restrict__ s,
                                                  float* __restrict__ c,
                                                  float* __restrict__ alpha) {
  int idx = blockIdx.x * 256 + threadIdx.x;
  if (idx < Bn * Hn) { s[idx] = 0.f; c[idx] = 0.f; }
  if (idx < Bn * Tn) alpha[idx] = 0.f;
}

// ---------------- enc_proj = hbatch(B*T x E) @ W_he(E x A) ----------------

__global__ __launch_bounds__(128) void encproj_gemm(const __bf16* __restrict__ Apk, // packed hbatch
                                                    const __bf16* __restrict__ Wt_he, // A x E
                                                    float* __restrict__ C) {          // B*T x A
  const int lane = threadIdx.x & 31;
  const int tile = blockIdx.x * 4 + (threadIdx.x >> 5);
  const int NT = An / 16;
  const int MT = (Bn * Tn) / 16;
  const int K32 = En / 32;
  if (tile >= MT * NT) return;
  const int mt = tile / NT, nt = tile - mt * NT;
  const int r16 = lane & 15;
  const int hl  = lane >> 4;
  const int col = nt * 16 + r16;
  v8f acc = {};
#pragma unroll 4
  for (int kb = 0; kb < K32; ++kb) {
    v16bf a = load_a_packed(Apk, mt * K32 + kb, lane);
    v16bf b = load_b_frag(Wt_he, En, col, kb * 32 + hl * 16);
    acc = wmma_bf16(a, b, acc);
  }
#pragma unroll
  for (int r = 0; r < 8; ++r)
    C[(size_t)(mt * 16 + r + hl * 8) * An + col] = acc[r];
}

// ---------------- sproj = s(16 x H) @ W_se(H x A) ----------------

__global__ __launch_bounds__(128) void sproj_gemm(const __bf16* __restrict__ spk,
                                                  const __bf16* __restrict__ Wt_se, // A x H
                                                  float* __restrict__ sproj) {      // 16 x A
  const int lane = threadIdx.x & 31;
  const int tile = blockIdx.x * 4 + (threadIdx.x >> 5);
  if (tile >= An / 16) return;
  const int r16 = lane & 15, hl = lane >> 4;
  const int col = tile * 16 + r16;
  v8f acc = {};
#pragma unroll 4
  for (int kb = 0; kb < Hn / 32; ++kb) {
    v16bf a = load_a_packed(spk, kb, lane);
    v16bf b = load_b_frag(Wt_se, Hn, col, kb * 32 + hl * 16);
    acc = wmma_bf16(a, b, acc);
  }
#pragma unroll
  for (int r = 0; r < 8; ++r)
    sproj[(size_t)(r + hl * 8) * An + col] = acc[r];
}

// ---------------- attention energies ----------------
// e[b][t] = w_att . tanh(enc_proj[b,t,:] + f_proj[b,t,:] + sproj[b,:] + b_att)

__global__ __launch_bounds__(256) void energy_kernel(const float* __restrict__ alpha,
                                                     const float* __restrict__ enc_proj,
                                                     const float* __restrict__ sproj,
                                                     const float* __restrict__ b_att,
                                                     const float* __restrict__ w_att,
                                                     const float* __restrict__ W_fe,   // CCH x A
                                                     const float* __restrict__ F_conv, // CCH x 1 x Kw
                                                     const int* __restrict__ lengths,
                                                     float* __restrict__ e) {
  __shared__ float sAl[256 + Kw - 1];
  __shared__ float sF[CCHn * Kw];
  __shared__ float sWfe[CCHn * An];
  __shared__ float sBatt[An];
  __shared__ float sWatt[An];
  __shared__ float sSp[An];

  const int tid = threadIdx.x;
  const int b   = blockIdx.x >> 1;          // T/256 = 2 chunks per batch row
  const int t0  = (blockIdx.x & 1) * 256;

  for (int i = tid; i < 256 + Kw - 1; i += 256) {
    int ti = t0 + i - (Kw / 2);
    sAl[i] = (ti >= 0 && ti < Tn) ? alpha[b * Tn + ti] : 0.f;
  }
  for (int i = tid; i < CCHn * Kw; i += 256) sF[i] = F_conv[i];
  for (int i = tid; i < CCHn * An; i += 256) sWfe[i] = W_fe[i];
  if (tid < An) {
    sBatt[tid] = b_att[tid];
    sWatt[tid] = w_att[tid];
    sSp[tid]   = sproj[b * An + tid];
  }
  __syncthreads();

  const int t = t0 + tid;
  float convf[CCHn];
#pragma unroll
  for (int ch = 0; ch < CCHn; ++ch) {
    float acc = 0.f;
    for (int k = 0; k < Kw; ++k) acc += sAl[tid + k] * sF[ch * Kw + k];
    convf[ch] = acc;
  }

  const float* ep = enc_proj + ((size_t)b * Tn + t) * An;
  float en = 0.f;
  for (int a = 0; a < An; ++a) {
    float v = ep[a] + sSp[a] + sBatt[a];
#pragma unroll
    for (int ch = 0; ch < CCHn; ++ch) v += convf[ch] * sWfe[ch * An + a];
    en += tanhf(v) * sWatt[a];
  }
  if (t >= lengths[b]) en = -1e30f;
  e[b * Tn + t] = en;
}

// ---------------- masked softmax over T, per batch row ----------------

__global__ __launch_bounds__(256) void softmax_kernel(const float* __restrict__ e,
                                                      float* __restrict__ alpha) {
  __shared__ float red[256];
  const int b = blockIdx.x, tid = threadIdx.x;
  float v0 = e[b * Tn + tid];
  float v1 = e[b * Tn + tid + 256];
  red[tid] = fmaxf(v0, v1);
  __syncthreads();
  for (int s2 = 128; s2 > 0; s2 >>= 1) {
    if (tid < s2) red[tid] = fmaxf(red[tid], red[tid + s2]);
    __syncthreads();
  }
  const float m = red[0];
  __syncthreads();
  float x0 = expf(v0 - m), x1 = expf(v1 - m);
  red[tid] = x0 + x1;
  __syncthreads();
  for (int s2 = 128; s2 > 0; s2 >>= 1) {
    if (tid < s2) red[tid] += red[tid + s2];
    __syncthreads();
  }
  const float inv = 1.f / red[0];
  alpha[b * Tn + tid]       = x0 * inv;
  alpha[b * Tn + tid + 256] = x1 * inv;
}

// ---------------- context g[b,e] = sum_t alpha[b,t] * h[b,t,e] ----------------

__global__ __launch_bounds__(256) void context_kernel(const float* __restrict__ alpha,
                                                      const float* __restrict__ h,
                                                      float* __restrict__ g) {
  __shared__ float sA[Tn];
  const int b  = blockIdx.x >> 2;
  const int ei = (blockIdx.x & 3) * 256 + threadIdx.x;
  for (int i = threadIdx.x; i < Tn; i += 256) sA[i] = alpha[b * Tn + i];
  __syncthreads();
  float acc = 0.f;
  const float* hp = h + (size_t)b * Tn * En + ei;
  for (int t = 0; t < Tn; ++t) acc += sA[t] * hp[(size_t)t * En];
  g[b * En + ei] = acc;
}

// ---------------- z = tanh(g @ W_gy + s @ W_sy + b_gy) ----------------

__global__ __launch_bounds__(128) void zhid_gemm(const __bf16* __restrict__ gpk,
                                                 const __bf16* __restrict__ spk,
                                                 const __bf16* __restrict__ Wt_gy, // H x E
                                                 const __bf16* __restrict__ Wt_sy, // H x H
                                                 const float* __restrict__ b_gy,
                                                 float* __restrict__ z) {
  const int lane = threadIdx.x & 31;
  const int tile = blockIdx.x * 4 + (threadIdx.x >> 5);
  if (tile >= Hn / 16) return;
  const int r16 = lane & 15, hl = lane >> 4;
  const int col = tile * 16 + r16;
  v8f acc = {};
#pragma unroll 4
  for (int kb = 0; kb < En / 32; ++kb) {
    v16bf a = load_a_packed(gpk, kb, lane);
    v16bf b = load_b_frag(Wt_gy, En, col, kb * 32 + hl * 16);
    acc = wmma_bf16(a, b, acc);
  }
#pragma unroll 4
  for (int kb = 0; kb < Hn / 32; ++kb) {
    v16bf a = load_a_packed(spk, kb, lane);
    v16bf b = load_b_frag(Wt_sy, Hn, col, kb * 32 + hl * 16);
    acc = wmma_bf16(a, b, acc);
  }
  const float bias = b_gy[col];
#pragma unroll
  for (int r = 0; r < 8; ++r)
    z[(size_t)(r + hl * 8) * Hn + col] = tanhf(acc[r] + bias);
}

// ---------------- out[:, step, :] = z @ W_yy + b_yy ----------------
// Packed A is async-staged into LDS once per workgroup (gfx1250 ASYNCcnt path),
// then all 4 waves consume it via LDS fragment reads.

__global__ __launch_bounds__(128) void y_gemm(const __bf16* __restrict__ Apk, // packed z (16 KB)
                                              const __bf16* __restrict__ Wt_yy, // V x H
                                              const float* __restrict__ b_yy,
                                              float* __restrict__ out, int step) {
  __shared__ __align__(32) __bf16 sA[(Hn / 32) * 32 * 16]; // 8192 bf16 = 16 KB

  {
    const uint32_t lbase = (uint32_t)(uintptr_t)&sA[0];
    const uint64_t gbase = (uint64_t)(uintptr_t)Apk;
#pragma unroll
    for (int i = 0; i < 8; ++i) {
      const int c = threadIdx.x + i * 128;          // 1024 chunks x 16 B
      uint32_t laddr = lbase + c * 16;
      uint64_t gaddr = gbase + (uint64_t)c * 16;
      asm volatile("global_load_async_to_lds_b128 %0, %1, off"
                   :: "v"(laddr), "v"(gaddr) : "memory");
    }
    asm volatile("s_wait_asynccnt 0" ::: "memory");
  }
  __syncthreads();

  const int lane = threadIdx.x & 31;
  const int tile = blockIdx.x * 4 + (threadIdx.x >> 5);
  if (tile >= Vn / 16) return;
  const int r16 = lane & 15, hl = lane >> 4;
  const int col = tile * 16 + r16;
  v8f acc = {};
#pragma unroll 4
  for (int kb = 0; kb < Hn / 32; ++kb) {
    v16bf a = *(const v16bf*)(&sA[(kb * 32 + lane) * 16]);
    v16bf b = load_b_frag(Wt_yy, Hn, col, kb * 32 + hl * 16);
    acc = wmma_bf16(a, b, acc);
  }
  const float bias = b_yy[col];
#pragma unroll
  for (int r = 0; r < 8; ++r) {
    int bi = r + hl * 8;
    out[(size_t)bi * Ln * Vn + (size_t)step * Vn + col] = acc[r] + bias;
  }
}

// ---------------- rec = E_ys[tgt] + s @ W_ss + g @ W_gs + b_gs ----------------

__global__ __launch_bounds__(128) void rec_gemm(const __bf16* __restrict__ spk,
                                                const __bf16* __restrict__ gpk,
                                                const __bf16* __restrict__ Wt_ss, // 4H x H
                                                const __bf16* __restrict__ Wt_gs, // 4H x E
                                                const float* __restrict__ E_ys,   // V x 4H
                                                const float* __restrict__ b_gs,
                                                const int* __restrict__ targets,  // B x L
                                                int step,
                                                float* __restrict__ rec) {
  const int lane = threadIdx.x & 31;
  const int tile = blockIdx.x * 4 + (threadIdx.x >> 5);
  if (tile >= H4 / 16) return;
  const int r16 = lane & 15, hl = lane >> 4;
  const int col = tile * 16 + r16;
  v8f acc = {};
#pragma unroll 4
  for (int kb = 0; kb < Hn / 32; ++kb) {
    v16bf a = load_a_packed(spk, kb, lane);
    v16bf b = load_b_frag(Wt_ss, Hn, col, kb * 32 + hl * 16);
    acc = wmma_bf16(a, b, acc);
  }
#pragma unroll 4
  for (int kb = 0; kb < En / 32; ++kb) {
    v16bf a = load_a_packed(gpk, kb, lane);
    v16bf b = load_b_frag(Wt_gs, En, col, kb * 32 + hl * 16);
    acc = wmma_bf16(a, b, acc);
  }
  const float bias = b_gs[col];
#pragma unroll
  for (int r = 0; r < 8; ++r) {
    int bi  = r + hl * 8;
    int tgt = targets[bi * Ln + step];
    rec[(size_t)bi * H4 + col] = acc[r] + E_ys[(size_t)tgt * H4 + col] + bias;
  }
}

// ---------------- LSTM pointwise update ----------------

__global__ __launch_bounds__(256) void lstm_kernel(const float* __restrict__ rec,
                                                   float* __restrict__ s,
                                                   float* __restrict__ c) {
  const int idx = blockIdx.x * 256 + threadIdx.x;  // B*H = 8192
  const int b = idx / Hn, j = idx - b * Hn;
  const float* r = rec + (size_t)b * H4;
  const float ig = r[j], fg = r[Hn + j], cg = r[2 * Hn + j], og = r[3 * Hn + j];
  const float cn = sigmoidf_(fg) * c[idx] + sigmoidf_(ig) * tanhf(cg);
  c[idx] = cn;
  s[idx] = sigmoidf_(og) * tanhf(cn);
}

// ---------------- host ----------------

extern "C" void kernel_launch(void* const* d_in, const int* in_sizes, int n_in,
                              void* d_out, int out_size, void* d_ws, size_t ws_size,
                              hipStream_t stream) {
  (void)in_sizes; (void)n_in; (void)out_size; (void)ws_size;

  const float* hbatch  = (const float*)d_in[0];
  const int*   lengths = (const int*)d_in[1];
  const int*   targets = (const int*)d_in[2];
  const float* W_sy    = (const float*)d_in[3];
  const float* W_gy    = (const float*)d_in[4];
  const float* b_gy    = (const float*)d_in[5];
  const float* W_yy    = (const float*)d_in[6];
  const float* b_yy    = (const float*)d_in[7];
  const float* E_ys    = (const float*)d_in[8];
  const float* W_ss    = (const float*)d_in[9];
  const float* W_gs    = (const float*)d_in[10];
  const float* b_gs    = (const float*)d_in[11];
  const float* W_se    = (const float*)d_in[12];
  const float* W_he    = (const float*)d_in[13];
  const float* W_fe    = (const float*)d_in[14];
  const float* b_att   = (const float*)d_in[15];
  const float* w_att   = (const float*)d_in[16];
  const float* F_conv  = (const float*)d_in[17];
  float* out = (float*)d_out;

  char* ws = (char*)d_ws;
  size_t off = 0;
  auto alloc = [&](size_t bytes) -> void* {
    void* p = ws + off;
    off = (off + bytes + 255) & ~(size_t)255;
    return p;
  };
  __bf16* Wt_he = (__bf16*)alloc((size_t)An * En * 2);
  __bf16* Wt_yy = (__bf16*)alloc((size_t)Vn * Hn * 2);
  __bf16* Wt_sy = (__bf16*)alloc((size_t)Hn * Hn * 2);
  __bf16* Wt_gy = (__bf16*)alloc((size_t)Hn * En * 2);
  __bf16* Wt_ss = (__bf16*)alloc((size_t)H4 * Hn * 2);
  __bf16* Wt_gs = (__bf16*)alloc((size_t)H4 * En * 2);
  __bf16* Wt_se = (__bf16*)alloc((size_t)An * Hn * 2);
  __bf16* hpk   = (__bf16*)alloc((size_t)Bn * Tn * En * 2); // packed hbatch
  __bf16* spk   = (__bf16*)alloc((size_t)Bn * Hn * 2);      // packed s
  __bf16* gpk   = (__bf16*)alloc((size_t)Bn * En * 2);      // packed g
  __bf16* zpk   = (__bf16*)alloc((size_t)Bn * Hn * 2);      // packed z
  float* enc_proj = (float*)alloc((size_t)Bn * Tn * An * 4);
  float* alpha    = (float*)alloc((size_t)Bn * Tn * 4);
  float* eng      = (float*)alloc((size_t)Bn * Tn * 4);
  float* s_st     = (float*)alloc((size_t)Bn * Hn * 4);
  float* c_st     = (float*)alloc((size_t)Bn * Hn * 4);
  float* g_ctx    = (float*)alloc((size_t)Bn * En * 4);
  float* z_hid    = (float*)alloc((size_t)Bn * Hn * 4);
  float* sproj    = (float*)alloc((size_t)Bn * An * 4);
  float* rec      = (float*)alloc((size_t)Bn * H4 * 4);

  auto cvt = [&](const float* in, __bf16* o, int K, int N) {
    int n = K * N;
    cvt_transpose_bf16<<<(n + 255) / 256, 256, 0, stream>>>(in, o, K, N);
  };
  cvt(W_he, Wt_he, En, An);
  cvt(W_yy, Wt_yy, Hn, Vn);
  cvt(W_sy, Wt_sy, Hn, Hn);
  cvt(W_gy, Wt_gy, En, Hn);
  cvt(W_ss, Wt_ss, Hn, H4);
  cvt(W_gs, Wt_gs, En, H4);
  cvt(W_se, Wt_se, Hn, An);

  init_state<<<(Bn * Hn + 255) / 256, 256, 0, stream>>>(s_st, c_st, alpha);

  // pack hbatch once (16.8 MB bf16), then step-invariant encoder projection
  pack_a_bf16<<<((Bn * Tn * En) + 255) / 256, 256, 0, stream>>>(hbatch, hpk, Bn * Tn, En);
  {
    int tiles = ((Bn * Tn) / 16) * (An / 16);
    encproj_gemm<<<(tiles + 3) / 4, 128, 0, stream>>>(hpk, Wt_he, enc_proj);
  }

  for (int step = 0; step < Ln; ++step) {
    pack_a_bf16<<<((Bn * Hn) + 255) / 256, 256, 0, stream>>>(s_st, spk, Bn, Hn);

    sproj_gemm<<<(An / 16 + 3) / 4, 128, 0, stream>>>(spk, Wt_se, sproj);

    energy_kernel<<<Bn * (Tn / 256), 256, 0, stream>>>(alpha, enc_proj, sproj, b_att,
                                                       w_att, W_fe, F_conv, lengths, eng);

    softmax_kernel<<<Bn, 256, 0, stream>>>(eng, alpha);

    context_kernel<<<Bn * (En / 256), 256, 0, stream>>>(alpha, hbatch, g_ctx);
    pack_a_bf16<<<((Bn * En) + 255) / 256, 256, 0, stream>>>(g_ctx, gpk, Bn, En);

    zhid_gemm<<<(Hn / 16 + 3) / 4, 128, 0, stream>>>(gpk, spk, Wt_gy, Wt_sy, b_gy, z_hid);
    pack_a_bf16<<<((Bn * Hn) + 255) / 256, 256, 0, stream>>>(z_hid, zpk, Bn, Hn);

    y_gemm<<<(Vn / 16 + 3) / 4, 128, 0, stream>>>(zpk, Wt_yy, b_yy, out, step);

    rec_gemm<<<(H4 / 16 + 3) / 4, 128, 0, stream>>>(spk, gpk, Wt_ss, Wt_gs, E_ys, b_gs,
                                                    targets, step, rec);

    lstm_kernel<<<(Bn * Hn) / 256, 256, 0, stream>>>(rec, s_st, c_st);
  }
}
